// VanillaRNN_60155311948065
// MI455X (gfx1250) — 16-bit and 8-bit matrix operand order repaired and checked by reading, compiled, not run on hardware
//
#include <hip/hip_runtime.h>
#include <math.h>

#define Bsz 64
#define Ssz 2048
#define Isz 256
#define Hsz 512

typedef __attribute__((ext_vector_type(16))) __bf16 v16bf;
typedef __attribute__((ext_vector_type(8)))  float  v8f;
typedef __attribute__((ext_vector_type(16))) int    v16i;
typedef __attribute__((ext_vector_type(4)))  unsigned int u32x4;

__device__ __forceinline__ unsigned int fp8_sw(float f) {
  unsigned u = __float_as_uint(f);
  unsigned s = (u >> 24) & 0x80u;
  unsigned a = u & 0x7fffffffu;
  if (a >= 0x7f800000u) return s | 0x7eu;
  int ef = (int)(a >> 23) - 127 + 7;
  unsigned m = (a >> 20) & 7u;
  unsigned r = (a >> 19) & 1u;
  if (ef <= 0) return s;
  unsigned v = ((unsigned)ef << 3) | m;
  v += r;
  if (v > 0x7eu) v = 0x7eu;
  return s | v;
}

__device__ __forceinline__ unsigned int fp8_byte(float f) {
#if __has_builtin(__builtin_amdgcn_cvt_pk_fp8_f32)
  return (unsigned)__builtin_amdgcn_cvt_pk_fp8_f32(f, 0.f, 0, false) & 0xffu;
#else
  return fp8_sw(f);
#endif
}

__device__ __forceinline__ unsigned int fp8x4(float a, float b, float c, float d) {
#if __has_builtin(__builtin_amdgcn_cvt_pk_fp8_f32)
  int v = 0;
  v = __builtin_amdgcn_cvt_pk_fp8_f32(a, b, v, false);
  v = __builtin_amdgcn_cvt_pk_fp8_f32(c, d, v, true);
  return (unsigned)v;
#else
  return fp8_sw(a) | (fp8_sw(b) << 8) | (fp8_sw(c) << 16) | (fp8_sw(d) << 24);
#endif
}

__global__ __launch_bounds__(256) void rnn_proj_kernel(
    const float* __restrict__ x, const float* __restrict__ V,
    const float* __restrict__ bias, float* __restrict__ out) {
  __shared__ __align__(16) __bf16 xt[16 * Isz];

  const int lane = threadIdx.x & 31;
  const int wave = threadIdx.x >> 5;
  const int row0 = blockIdx.x * 16;

  const float* xsrc = x + (size_t)row0 * Isz;
  for (int i = threadIdx.x; i < 16 * Isz; i += 256) xt[i] = (__bf16)xsrc[i];
  __syncthreads();

  const int hi   = lane >> 4;
  const int nloc = lane & 15;
  const int col0 = wave * 64;

  v8f acc[4] = {v8f{0}, v8f{0}, v8f{0}, v8f{0}};

  for (int kk = 0; kk < Isz; kk += 32) {
    union { v16bf v; u32x4 q[2]; } au;
    const __bf16* ap = xt + nloc * Isz + kk + hi * 8;
    au.q[0] = *(const u32x4*)(ap);
    au.q[1] = *(const u32x4*)(ap + 16);

#pragma unroll
    for (int tt = 0; tt < 4; ++tt) {
      const int col = col0 + tt * 16 + nloc;
      const float* vp = V + (size_t)col * Isz + kk + hi * 8;
      v16bf bf;
#pragma unroll
      for (int i = 0; i < 16; ++i) bf[i] = (__bf16)vp[i < 8 ? i : i + 8];
      acc[tt] = __builtin_amdgcn_wmma_f32_16x16x32_bf16(
          false, au.v, false, bf, (short)0, acc[tt], false, false);
    }
  }

#pragma unroll
  for (int tt = 0; tt < 4; ++tt) {
    const int col = col0 + tt * 16 + nloc;
    const float bv = bias[col];
#pragma unroll
    for (int r = 0; r < 8; ++r) {
      const int m = r + hi * 8;
      out[(size_t)(row0 + m) * Hsz + col] = acc[tt][r] + bv;
    }
  }
}

__global__ __launch_bounds__(512) void rnn_scan_kernel(
    const float* __restrict__ h0, const float* __restrict__ W,
    float* __restrict__ out  ,
    float* __restrict__ hn) {
  __shared__ __align__(16) unsigned char hbuf[2][16 * Hsz];

  const int lane = threadIdx.x & 31;
  const int wave = threadIdx.x >> 5;
  const int b0   = blockIdx.x * 16;
  const int hi   = lane >> 4;
  const int nloc = lane & 15;
  const int col0 = wave * 32;

  v16i Bw[2][4];
#pragma unroll
  for (int tt = 0; tt < 2; ++tt) {
    const int j = col0 + tt * 16 + nloc;
#pragma unroll
    for (int kki = 0; kki < 4; ++kki) {
#pragma unroll
      for (int c = 0; c < 4; ++c) {
        const float* wp = W + (size_t)j * Hsz + kki * 128 + hi * 16 + c * 32;
#pragma unroll
        for (int q = 0; q < 4; ++q) {
          Bw[tt][kki][4 * c + q] = (int)fp8x4(wp[4 * q + 0], wp[4 * q + 1],
                                              wp[4 * q + 2], wp[4 * q + 3]);
        }
      }
    }
  }

  for (int i = threadIdx.x; i < 16 * Hsz; i += 512) {
    const int m = i >> 9, k = i & (Hsz - 1);
    hbuf[0][i] = (unsigned char)fp8_byte(h0[(size_t)(b0 + m) * Hsz + k]);
  }
  __syncthreads();

  for (int t = 0; t < Ssz; ++t) {
    const unsigned char* hc = hbuf[t & 1];
    unsigned char* hnx      = hbuf[(t & 1) ^ 1];

    v8f acc[2];
#pragma unroll
    for (int tt = 0; tt < 2; ++tt) {
      const int col = col0 + tt * 16 + nloc;
#pragma unroll
      for (int r = 0; r < 8; ++r) {
        const int m = r + hi * 8;
        acc[tt][r] = out[((size_t)(b0 + m) * Ssz + t) * Hsz + col];
      }
    }

#pragma unroll
    for (int kki = 0; kki < 4; ++kki) {
      v16i afr;
      const unsigned char* ap = hc + nloc * Hsz + kki * 128 + hi * 16;
#pragma unroll
      for (int c = 0; c < 8; ++c) {
        afr[2 * c]     = *(const int*)(ap + (c >> 1) * 32 + (c & 1) * 8);
        afr[2 * c + 1] = *(const int*)(ap + (c >> 1) * 32 + (c & 1) * 8 + 4);
      }
#pragma unroll
      for (int tt = 0; tt < 2; ++tt) {
        acc[tt] = __builtin_amdgcn_wmma_f32_16x16x128_fp8_fp8(
            afr, Bw[tt][kki], (short)0, acc[tt], false, false);
      }
    }

#pragma unroll
    for (int tt = 0; tt < 2; ++tt) {
      const int col = col0 + tt * 16 + nloc;
#pragma unroll
      for (int r = 0; r < 8; ++r) {
        const int m  = r + hi * 8;
        const float hv = tanhf(acc[tt][r]);
        out[((size_t)(b0 + m) * Ssz + t) * Hsz + col] = hv;
        hnx[m * Hsz + col] = (unsigned char)fp8_byte(hv);
        if (t == Ssz - 1) hn[(size_t)(b0 + m) * Hsz + col] = hv;
      }
    }
    __syncthreads();
  }
}

extern "C" void kernel_launch(void* const* d_in, const int* in_sizes, int n_in,
                              void* d_out, int out_size, void* d_ws, size_t ws_size,
                              hipStream_t stream) {
  const float* x    = (const float*)d_in[0];
  const float* h0   = (const float*)d_in[1];
  const float* W    = (const float*)d_in[2];
  const float* V    = (const float*)d_in[3];
  const float* bias = (const float*)d_in[4];

  float* out = (float*)d_out;
  float* hn  = out + (size_t)Bsz * Ssz * Hsz;

  rnn_proj_kernel<<<dim3((Bsz * Ssz) / 16), dim3(256), 0, stream>>>(x, V, bias, out);

  rnn_scan_kernel<<<dim3(Bsz / 16), dim3(512), 0, stream>>>(h0, W, out, hn);
}
